// TiSASRec_27754078667279
// MI455X (gfx1250) — matrix operand order repaired and checked by reading, not compiled, not run on hardware
//
#include <hip/hip_runtime.h>

// ---------------- problem constants ----------------
#define Hn   128
#define Bn   16
#define Ln   200
#define NHn  2
#define NBn  2
#define Dn   64            // head dim
#define MLn  (Bn*Ln)       // 3200 rows
#define MROW 3232          // padded row count (key tiles reach row b*L+223)
#define QTLD 272           // padded 257 time buckets for QT matrix
#define TPAD 288           // padded bucket count, multiple of 32
#define NEGF (-4294967295.0f)

typedef __attribute__((ext_vector_type(16))) _Float16 v16h;
typedef __attribute__((ext_vector_type(8)))  float    v8f;

// ---------------------------------------------------------------------------
// WMMA GEMM, 16x64 tile per wave (4 accumulators share one A fragment):
//   C(M,N) = act( A(M,K) @ W(N,K)^T + bias + addrow )
//   B-matrix element B[k][n] = W[n*ldw + wcol + k]  (contiguous along k)
//   grid = (M/16, ceil(N/64)), block = 32 (one wave)
//   Optional outputs: C (f32), Ch (f16 row-major), ChT (f16 transposed).
// ---------------------------------------------------------------------------
__global__ void k_wmma_gemm(const _Float16* __restrict__ A, int lda, int acol,
                            const _Float16* __restrict__ W, int ldw, int wcol,
                            const float* __restrict__ bias,
                            const float* __restrict__ addrow, int addld, int rowmod,
                            float* __restrict__ C, int ldc,
                            _Float16* __restrict__ Ch, int ldch,
                            _Float16* __restrict__ ChT, int ldct,
                            int N, int K, int relu)
{
  const int lane = threadIdx.x & 31;
  const int mlo  = lane & 15;
  const int half = lane >> 4;
  const int m = blockIdx.x * 16 + mlo;            // A-frag row

  v8f acc[4] = {};
  for (int ks = 0; ks < K; ks += 32) {
    // A fragment: lane row m, K chunk = [ks+half*16,+16)
    v16h a;
    const _Float16* p0 = A + (size_t)m * lda + acol + ks + half * 16;
#pragma unroll
    for (int i = 0; i < 8; ++i) { a[i] = p0[i]; a[8 + i] = p0[8 + i]; }
#pragma unroll
    for (int t = 0; t < 4; ++t) {                 // 4 N-tiles reuse the A frag
      const int n  = blockIdx.y * 64 + t * 16 + mlo;
      const int nB = (n < N) ? n : (N - 1);       // clamp for padded N tiles
      v16h bf;
      const _Float16* q0 = W + (size_t)nB * ldw + wcol + ks + half * 16;
#pragma unroll
      for (int i = 0; i < 16; ++i) bf[i] = q0[i];
      acc[t] = __builtin_amdgcn_wmma_f32_16x16x32_f16(false, a, false, bf,
                                                      (short)0, acc[t], false, false);
    }
  }

#pragma unroll
  for (int t = 0; t < 4; ++t) {
    const int n = blockIdx.y * 64 + t * 16 + mlo;
    if (n < N) {
      const float bv = bias ? bias[n] : 0.0f;
#pragma unroll
      for (int r = 0; r < 8; ++r) {
        const int mm = blockIdx.x * 16 + half * 8 + r;   // D row for VGPR r
        float v = acc[t][r] + bv;
        if (addrow) v += addrow[(size_t)(mm % rowmod) * addld + n];
        if (relu)  v = fmaxf(v, 0.0f);
        if (C)   C  [(size_t)mm * ldc  + n]  = v;
        if (Ch)  Ch [(size_t)mm * ldch + n]  = (_Float16)v;
        if (ChT) ChT[(size_t)n  * ldct + mm] = (_Float16)v;
      }
    }
  }
}

// ---------------------------------------------------------------------------
// Attention: one wave per (query-tile, head, batch).
//   scores = Q@K'^T (WMMA) + gather(QT, tm)  -> mask -> softmax
//   out    = A@V'^T (WMMA, transposed V)  +  bucketed(A) @ timeV^T (WMMA)
// ---------------------------------------------------------------------------
__global__ void k_attn(const _Float16* __restrict__ Qh,     // [MROW][H]
                       const _Float16* __restrict__ Kh,     // [MROW][H]
                       const _Float16* __restrict__ VhT,    // [H][MROW] transposed
                       const float*    __restrict__ QT,     // [NH][MROW][QTLD]
                       const _Float16* __restrict__ TV16T,  // [H][TPAD] transposed
                       const int*      __restrict__ tmat,   // (B,L,L)
                       const int*      __restrict__ logseq, // (B,L)
                       float*          __restrict__ attn)   // [MROW][H]
{
  const int qt = blockIdx.x, h = blockIdx.y, b = blockIdx.z;
  const int lane = threadIdx.x & 31;
  const int mlo  = lane & 15;
  const int half = lane >> 4;
  const int rowbase = b * Ln + qt * 16;

  __shared__ float    sS[16][224];   // scores, then probabilities (f32)
  __shared__ _Float16 sA[16][224];   // probabilities (f16, A-matrix for WMMA)
  __shared__ float    sW[16][TPAD];  // per-row time-bucket weights

  // ---- phase 1: raw scores Q @ K'^T (Q fragments hoisted over key tiles) ----
  v16h qa0, qa1;
  {
    const _Float16* p0 = Qh + (size_t)(rowbase + mlo) * Hn + h * Dn + half * 16;
#pragma unroll
    for (int i = 0; i < 8; ++i) { qa0[i] = p0[i];      qa0[8 + i] = p0[8 + i];
                                  qa1[i] = p0[32 + i]; qa1[8 + i] = p0[40 + i]; }
  }
  for (int nt = 0; nt < 13; ++nt) {
    const _Float16* q0 = Kh + (size_t)(b * Ln + nt * 16 + mlo) * Hn + h * Dn + half * 16;
    v16h bf0, bf1;
#pragma unroll
    for (int i = 0; i < 16; ++i) { bf0[i] = q0[i]; bf1[i] = q0[32 + i]; }
    v8f acc = {};
    acc = __builtin_amdgcn_wmma_f32_16x16x32_f16(false, qa0, false, bf0,
                                                 (short)0, acc, false, false);
    acc = __builtin_amdgcn_wmma_f32_16x16x32_f16(false, qa1, false, bf1,
                                                 (short)0, acc, false, false);
#pragma unroll
    for (int r = 0; r < 8; ++r) sS[half * 8 + r][nt * 16 + mlo] = acc[r];
  }
  __syncthreads();

  // ---- phase 2: + time-key term (gather), scale, mask ----
  for (int q = 0; q < 16; ++q) {
    const int qg = qt * 16 + q;
    const bool rowpad = (qg >= Ln) || (logseq[b * Ln + qg] == 0);
    const float* qtrow = QT + (size_t)h * MROW * QTLD + (size_t)(b * Ln + qg) * QTLD;
    for (int k = lane; k < 224; k += 32) {
      float s = NEGF;
      if (!rowpad && qg < Ln && k < Ln && k <= qg) {
        const int t = tmat[((size_t)(b * Ln + qg)) * Ln + k];
        s = (sS[q][k] + qtrow[t]) * 0.125f;   // 1/sqrt(64)
      }
      sS[q][k] = s;
    }
  }
  __syncthreads();

  // ---- phase 3: row softmax (one lane per query row) ----
  if (lane < 16) {
    const int q = lane;
    float mx = -3.4e38f;
    for (int k = 0; k < Ln; ++k) mx = fmaxf(mx, sS[q][k]);
    float sum = 0.0f;
    for (int k = 0; k < Ln; ++k) sum += __expf(sS[q][k] - mx);
    const float inv = 1.0f / sum;
    for (int k = 0; k < 224; ++k) {
      const float p = (k < Ln) ? __expf(sS[q][k] - mx) * inv : 0.0f;
      sS[q][k] = p;
      sA[q][k] = (_Float16)p;
    }
  }
  __syncthreads();

  // ---- phase 4: scatter probs into time buckets (LDS f32 atomics) ----
  for (int i = lane; i < 16 * TPAD; i += 32) (&sW[0][0])[i] = 0.0f;
  __syncthreads();
  for (int i = lane; i < 16 * Ln; i += 32) {
    const int q = i / Ln, k = i % Ln;
    const int qg = qt * 16 + q;
    if (qg < Ln) {
      const float p = sS[q][k];
      if (p != 0.0f) {
        const int t = tmat[((size_t)(b * Ln + qg)) * Ln + k];
        atomicAdd(&sW[q][t], p);
      }
    }
  }
  __syncthreads();

  // ---- phase 5: out = A@V'^T + buckets@timeV^T (4 d-tiles share A frag) ----
  v8f acc4[4] = {};
  for (int ks = 0; ks < 224; ks += 32) {           // keys
    v16h a;
    const _Float16* p0 = &sA[mlo][ks + half * 16];
#pragma unroll
    for (int i = 0; i < 8; ++i) { a[i] = p0[i]; a[8 + i] = p0[8 + i]; }
#pragma unroll
    for (int t = 0; t < 4; ++t) {
      const int col = h * Dn + t * 16 + mlo;
      v16h bf;
      const _Float16* vp = VhT + (size_t)col * MROW + b * Ln + ks + half * 16;
#pragma unroll
      for (int i = 0; i < 16; ++i) bf[i] = vp[i];  // contiguous
      acc4[t] = __builtin_amdgcn_wmma_f32_16x16x32_f16(false, a, false, bf,
                                                       (short)0, acc4[t], false, false);
    }
  }
  for (int ks = 0; ks < TPAD; ks += 32) {          // time buckets
    v16h a;
    const float* p0 = &sW[mlo][ks + half * 16];
#pragma unroll
    for (int i = 0; i < 8; ++i) { a[i] = (_Float16)p0[i]; a[8 + i] = (_Float16)p0[8 + i]; }
#pragma unroll
    for (int t = 0; t < 4; ++t) {
      const int col = h * Dn + t * 16 + mlo;
      v16h bf;
      const _Float16* tp = TV16T + (size_t)col * TPAD + ks + half * 16;
#pragma unroll
      for (int i = 0; i < 16; ++i) bf[i] = tp[i];  // contiguous
      acc4[t] = __builtin_amdgcn_wmma_f32_16x16x32_f16(false, a, false, bf,
                                                       (short)0, acc4[t], false, false);
    }
  }
#pragma unroll
  for (int t = 0; t < 4; ++t) {
#pragma unroll
    for (int r = 0; r < 8; ++r) {
      const int mrow = qt * 16 + half * 8 + r;
      if (mrow < Ln)
        attn[(size_t)(b * Ln + mrow) * Hn + h * Dn + t * 16 + mlo] = acc4[t][r];
    }
  }
}

// ---------------------------------------------------------------------------
// LayerNorm (+optional residual, +optional keep-mask). One 128-thread block/row.
// ---------------------------------------------------------------------------
__global__ void k_ln(const float* __restrict__ x, const float* __restrict__ res,
                     const float* __restrict__ g, const float* __restrict__ bb,
                     const int* __restrict__ logseq,
                     float* __restrict__ out, _Float16* __restrict__ outh)
{
  const int row = blockIdx.x, t = threadIdx.x;
  __shared__ float red[Hn];
  float v = x[(size_t)row * Hn + t] + (res ? res[(size_t)row * Hn + t] : 0.0f);
  red[t] = v; __syncthreads();
  for (int s = Hn / 2; s > 0; s >>= 1) { if (t < s) red[t] += red[t + s]; __syncthreads(); }
  const float mean = red[0] / (float)Hn;
  __syncthreads();
  const float dv = v - mean;
  red[t] = dv * dv; __syncthreads();
  for (int s = Hn / 2; s > 0; s >>= 1) { if (t < s) red[t] += red[t + s]; __syncthreads(); }
  const float var = red[0] / (float)Hn;
  float y = dv * rsqrtf(var + 1e-8f) * g[t] + bb[t];
  if (logseq && logseq[row] == 0) y = 0.0f;
  out[(size_t)row * Hn + t] = y;
  if (outh) outh[(size_t)row * Hn + t] = (_Float16)y;
}

// embedding gather: seqs = item_emb[log_seqs]*sqrt(H)*keep
__global__ void k_gather(const float* __restrict__ item_emb,
                         const int* __restrict__ logseq,
                         float* __restrict__ seqs, _Float16* __restrict__ seqs16)
{
  const int row = blockIdx.x, t = threadIdx.x;
  const int idx = logseq[row];
  float v = item_emb[(size_t)idx * Hn + t] * 11.313708498984761f;  // sqrt(128)
  if (idx == 0) v = 0.0f;
  seqs  [(size_t)row * Hn + t] = v;
  seqs16[(size_t)row * Hn + t] = (_Float16)v;
}

__global__ void k_cvt(const float* __restrict__ src, _Float16* __restrict__ dst, int n)
{
  for (int i = blockIdx.x * blockDim.x + threadIdx.x; i < n; i += gridDim.x * blockDim.x)
    dst[i] = (_Float16)src[i];
}

// transposed f16 convert: dst[c*ldt + r] = src[r*cols + c]   (r<rows, c<cols)
__global__ void k_cvtT(const float* __restrict__ src, _Float16* __restrict__ dst,
                       int rows, int cols, int ldt)
{
  for (int i = blockIdx.x * blockDim.x + threadIdx.x; i < rows * cols;
       i += gridDim.x * blockDim.x) {
    const int r = i / cols, c = i % cols;
    dst[(size_t)c * ldt + r] = (_Float16)src[i];
  }
}

// pos/neg logits: dot(feats[row], item_emb[idx]) -> out[y*ML + row]
__global__ void k_logits(const float* __restrict__ feats, const float* __restrict__ item_emb,
                         const int* __restrict__ poss, const int* __restrict__ negs,
                         float* __restrict__ out)
{
  const int row = blockIdx.x, t = threadIdx.x;
  const int idx = (blockIdx.y == 0) ? poss[row] : negs[row];
  __shared__ float red[Hn];
  red[t] = feats[(size_t)row * Hn + t] * item_emb[(size_t)idx * Hn + t];
  __syncthreads();
  for (int s = Hn / 2; s > 0; s >>= 1) { if (t < s) red[t] += red[t + s]; __syncthreads(); }
  if (t == 0) out[(size_t)blockIdx.y * MLn + row] = red[0];
}

// ---------------------------------------------------------------------------
extern "C" void kernel_launch(void* const* d_in, const int* in_sizes, int n_in,
                              void* d_out, int out_size, void* d_ws, size_t ws_size,
                              hipStream_t stream) {
  if (n_in < 26) return;
  const int*   logseq  = (const int*)  d_in[1];
  const int*   tmat    = (const int*)  d_in[2];
  const int*   poss    = (const int*)  d_in[3];
  const int*   negs    = (const int*)  d_in[4];
  const float* itemE   = (const float*)d_in[5];
  const float* posK    = (const float*)d_in[6];
  const float* posV    = (const float*)d_in[7];
  const float* timeK   = (const float*)d_in[8];
  const float* timeV   = (const float*)d_in[9];
  const float* Wq = (const float*)d_in[10]; const float* bq = (const float*)d_in[11];
  const float* Wk = (const float*)d_in[12]; const float* bk = (const float*)d_in[13];
  const float* Wv = (const float*)d_in[14]; const float* bv = (const float*)d_in[15];
  const float* g1 = (const float*)d_in[16]; const float* b1l = (const float*)d_in[17];
  const float* W1 = (const float*)d_in[18]; const float* b1f = (const float*)d_in[19];
  const float* W2 = (const float*)d_in[20]; const float* b2f = (const float*)d_in[21];
  const float* g2 = (const float*)d_in[22]; const float* b2l = (const float*)d_in[23];
  const float* gf = (const float*)d_in[24]; const float* bfl = (const float*)d_in[25];
  float* out = (float*)d_out;

  // ---- workspace carve-up ----
  char* ws = (char*)d_ws;
  size_t off = 0;
  auto carve = [&](size_t bytes) { size_t o = off; off = (off + bytes + 255) & ~(size_t)255; return o; };
  const size_t RH4 = (size_t)MROW * Hn * 4, RH2 = (size_t)MROW * Hn * 2;
  const size_t o_seqs   = carve(RH4);
  const size_t o_seqs16 = carve(RH2);
  const size_t o_Qh     = carve(RH2);
  const size_t o_Kh     = carve(RH2);
  const size_t o_VhT    = carve(RH2);                       // [H][MROW]
  const size_t o_hid16  = carve(RH2);
  const size_t o_attn   = carve(RH4);
  const size_t o_ffn    = carve(RH4);
  const size_t o_feats  = carve(RH4);
  const size_t o_QT     = carve((size_t)NHn * MROW * QTLD * 4);
  const size_t o_Wq16   = carve((size_t)NBn * Hn * Hn * 2);
  const size_t o_Wk16   = carve((size_t)NBn * Hn * Hn * 2);
  const size_t o_Wv16   = carve((size_t)NBn * Hn * Hn * 2);
  const size_t o_W116   = carve((size_t)NBn * Hn * Hn * 2);
  const size_t o_W216   = carve((size_t)NBn * Hn * Hn * 2);
  const size_t o_TK16   = carve((size_t)TPAD * Hn * 2);     // [TPAD][H]
  const size_t o_TV16T  = carve((size_t)Hn * TPAD * 2);     // [H][TPAD]
  if (off > ws_size) return;

  float*    seqs   = (float*)   (ws + o_seqs);
  _Float16* seqs16 = (_Float16*)(ws + o_seqs16);
  _Float16* Qh     = (_Float16*)(ws + o_Qh);
  _Float16* Kh     = (_Float16*)(ws + o_Kh);
  _Float16* VhT    = (_Float16*)(ws + o_VhT);
  _Float16* hid16  = (_Float16*)(ws + o_hid16);
  float*    attn   = (float*)   (ws + o_attn);
  float*    ffn    = (float*)   (ws + o_ffn);
  float*    feats  = (float*)   (ws + o_feats);
  float*    QT     = (float*)   (ws + o_QT);
  _Float16* Wq16   = (_Float16*)(ws + o_Wq16);
  _Float16* Wk16   = (_Float16*)(ws + o_Wk16);
  _Float16* Wv16   = (_Float16*)(ws + o_Wv16);
  _Float16* W116   = (_Float16*)(ws + o_W116);
  _Float16* W216   = (_Float16*)(ws + o_W216);
  _Float16* TK16   = (_Float16*)(ws + o_TK16);
  _Float16* TV16T  = (_Float16*)(ws + o_TV16T);

  // zero all scratch (pads must be 0 so padded WMMA lanes see finite zeros)
  hipMemsetAsync(d_ws, 0, off, stream);

  const int HH = Hn * Hn;
  const dim3 blk32(32), blk128(Hn), blk256(256);
  // f16 copies of weights and time embeddings
  k_cvt<<<64, blk256, 0, stream>>>(Wq, Wq16, NBn * HH);
  k_cvt<<<64, blk256, 0, stream>>>(Wk, Wk16, NBn * HH);
  k_cvt<<<64, blk256, 0, stream>>>(Wv, Wv16, NBn * HH);
  k_cvt<<<64, blk256, 0, stream>>>(W1, W116, NBn * HH);
  k_cvt<<<64, blk256, 0, stream>>>(W2, W216, NBn * HH);
  k_cvt <<<64, blk256, 0, stream>>>(timeK, TK16, 257 * Hn);
  k_cvtT<<<64, blk256, 0, stream>>>(timeV, TV16T, 257, Hn, TPAD);

  k_gather<<<MLn, blk128, 0, stream>>>(itemE, logseq, seqs, seqs16);

  const dim3 gemmHH(MLn / 16, Hn / 64);         // M=3200, N=128 -> (200, 2)
  const dim3 gemmQT(MLn / 16, (257 + 63) / 64); // N=257 -> (200, 5)
  for (int i = 0; i < NBn; ++i) {
    // Q = seqs @ Wq^T + bq          (f16 row-major copy)
    k_wmma_gemm<<<gemmHH, blk32, 0, stream>>>(seqs16, Hn, 0, Wq16 + i * HH, Hn, 0,
        bq + i * Hn, nullptr, 0, 1, nullptr, 0, Qh, Hn, nullptr, 0, Hn, Hn, 0);
    // K' = seqs @ Wk^T + bk + posK  (pos emb folded in)
    k_wmma_gemm<<<gemmHH, blk32, 0, stream>>>(seqs16, Hn, 0, Wk16 + i * HH, Hn, 0,
        bk + i * Hn, posK, Hn, Ln, nullptr, 0, Kh, Hn, nullptr, 0, Hn, Hn, 0);
    // V' = seqs @ Wv^T + bv + posV  (f16 TRANSPOSED copy for A@V WMMA)
    k_wmma_gemm<<<gemmHH, blk32, 0, stream>>>(seqs16, Hn, 0, Wv16 + i * HH, Hn, 0,
        bv + i * Hn, posV, Hn, Ln, nullptr, 0, nullptr, 0, VhT, MROW, Hn, Hn, 0);
    // QT[h] = Q_head @ timeK_head^T  (M=3200, N=257, K=64)
    for (int h = 0; h < NHn; ++h)
      k_wmma_gemm<<<gemmQT, blk32, 0, stream>>>(Qh, Hn, h * Dn, TK16, Hn, h * Dn,
          nullptr, nullptr, 0, 1, QT + (size_t)h * MROW * QTLD, QTLD,
          nullptr, 0, nullptr, 0, 257, Dn, 0);
    // attention (scores + softmax + A@V' + time buckets @ timeV)
    k_attn<<<dim3(13, NHn, Bn), blk32, 0, stream>>>(Qh, Kh, VhT, QT, TV16T,
                                                    tmat, logseq, attn);
    // seqs = LN(seqs + attn)
    k_ln<<<MLn, blk128, 0, stream>>>(seqs, attn, g1 + i * Hn, b1l + i * Hn,
                                     nullptr, seqs, seqs16);
    // hid = relu(seqs @ W1^T + b1)
    k_wmma_gemm<<<gemmHH, blk32, 0, stream>>>(seqs16, Hn, 0, W116 + i * HH, Hn, 0,
        b1f + i * Hn, nullptr, 0, 1, nullptr, 0, hid16, Hn, nullptr, 0, Hn, Hn, 1);
    // ffn = hid @ W2^T + b2
    k_wmma_gemm<<<gemmHH, blk32, 0, stream>>>(hid16, Hn, 0, W216 + i * HH, Hn, 0,
        b2f + i * Hn, nullptr, 0, 1, ffn, Hn, nullptr, 0, nullptr, 0, Hn, Hn, 0);
    // seqs = LN(seqs + ffn) * keep
    k_ln<<<MLn, blk128, 0, stream>>>(seqs, ffn, g2 + i * Hn, b2l + i * Hn,
                                     logseq, seqs, seqs16);
  }
  // final layernorm + logits
  k_ln<<<MLn, blk128, 0, stream>>>(seqs, nullptr, gf, bfl, nullptr, feats, nullptr);
  k_logits<<<dim3(MLn, 2), blk128, 0, stream>>>(feats, itemE, poss, negs, out);
  (void)in_sizes; (void)out_size;
}
